// Seq2Seq_16913581212001
// MI455X (gfx1250) — matrix operand order repaired and checked by reading, not compiled, not run on hardware
//
#include <hip/hip_runtime.h>
#include <hip/hip_bf16.h>

// ---------------------------------------------------------------------------
// Seq2Seq (GRU encoder + dot-attention GRU decoder + vocab projection)
// CDNA5 / gfx1250: wave32, WMMA f32<-f16 16x16x32 tensor path.
//
// Strategy:
//   * Hoist all batched GEMMs out of the sequential scans:
//       enc_gi  = embed(src) @ enc_Wih^T + bih          [8192 x 1536]  (WMMA)
//       dec_gie = embed(tgt) @ dec_Wih[:, :E]^T + bih   [2048 x 1536]  (WMMA)
//       logits  = pred @ fc_W^T + fc_b                  [2048 x 32000] (WMMA, dominant)
//     Batched GEMM wave tile = 32M x 64N: 8 WMMAs per 192 B/lane loaded
//     (~21 FLOP/byte from L0) so the tensor pipe is not load-starved.
//   * Sequential recurrence per step uses small WMMA GEMMs (M=32):
//       gh  = h   @ Whh^T    [32 x 1536]
//       gic = ctx @ WihC^T   [32 x 1536]   (decoder only)
//     followed by a pointwise gate kernel (f32).
//   * h kept f32 across steps; only GEMM operands rounded to f16 (f32 accum).
// ---------------------------------------------------------------------------

typedef _Float16 half_t;
typedef __attribute__((ext_vector_type(16))) _Float16 v16h;
typedef __attribute__((ext_vector_type(8)))  _Float16 v8h;
typedef __attribute__((ext_vector_type(8)))  float    v8f;

#define SS_B 32
#define SS_S 256
#define SS_T 64
#define SS_E 256
#define SS_H 512
#define SS_V 32000
#define SS_G 1536   // 3*H

__device__ __forceinline__ v16h ss_load_afrag(const half_t* ap) {
    // 16-bit A 16x32 layout: elems 0..15 -> K = kh*16 + i (contiguous, matches B)
    v8h alo = *(const v8h*)(ap);
    v8h ahi = *(const v8h*)(ap + 8);
    return __builtin_shufflevector(alo, ahi,
        0,1,2,3,4,5,6,7,8,9,10,11,12,13,14,15);
}

// ---------------------------------------------------------------------------
// Batched WMMA GEMM:  C[M,N] = A[M,K] @ W[N,K]^T + bias[N]
// block = 128 threads = 4 waves; wave tile = 32(M) x 64(N); block tile = 128 x 64
// Requires M%128==0, N%64==0, K%32==0 (true for all call sites).
// Per k-step each lane loads 2 A-frags (64B) + 4 B-frags (128B) feeding 8 WMMAs.
// ---------------------------------------------------------------------------
__global__ __launch_bounds__(128) void ss_gemm_wmma(
    const half_t* __restrict__ A, const half_t* __restrict__ W,
    const float* __restrict__ bias, float* __restrict__ C,
    int M, int N, int K)
{
    const int lane = threadIdx.x & 31;
    const int wave = threadIdx.x >> 5;
    const int n0 = blockIdx.x * 64;
    const int m0 = (blockIdx.y * 4 + wave) * 32;
    if (m0 >= M) return;                    // uniform per wave

    const int lm = lane & 15;               // M row within tile (A) / N col (B,C)
    const int kh = lane >> 4;               // K-half selector

    v8f acc0[4] = {};
    v8f acc1[4] = {};
    const half_t* arow0 = A + (size_t)(m0 + lm) * K;
    const half_t* arow1 = A + (size_t)(m0 + 16 + lm) * K;

    for (int k0 = 0; k0 < K; k0 += 32) {
        v16h afrag0 = ss_load_afrag(arow0 + k0 + kh * 16);
        v16h afrag1 = ss_load_afrag(arow1 + k0 + kh * 16);
#pragma unroll
        for (int t = 0; t < 4; ++t) {
            const half_t* bp = W + (size_t)(n0 + t * 16 + lm) * K + k0 + kh * 16;
            v16h bfrag = *(const v16h*)bp;
            acc0[t] = __builtin_amdgcn_wmma_f32_16x16x32_f16(
                false, afrag0, false, bfrag, (short)0, acc0[t], false, false);
            acc1[t] = __builtin_amdgcn_wmma_f32_16x16x32_f16(
                false, afrag1, false, bfrag, (short)0, acc1[t], false, false);
        }
    }
#pragma unroll
    for (int t = 0; t < 4; ++t) {
        int n = n0 + t * 16 + lm;
        float bv = bias ? bias[n] : 0.0f;
#pragma unroll
        for (int i = 0; i < 8; ++i) {
            int m = m0 + kh * 8 + i;
            C[(size_t)m * N + n]        = acc0[t][i] + bv;
            C[(size_t)(m + 16) * N + n] = acc1[t][i] + bv;
        }
    }
}

// ---------------------------------------------------------------------------
// Small-M recurrent WMMA GEMM: C[32,N] = A[32,512] @ W[N,512]^T
// block = 64 threads (2 waves, one per M-tile); grid.x = N/64. Latency-bound.
// ---------------------------------------------------------------------------
__global__ __launch_bounds__(64) void ss_gemm32_wmma(
    const half_t* __restrict__ A, const half_t* __restrict__ W,
    float* __restrict__ C, int N)
{
    const int lane = threadIdx.x & 31;
    const int wave = threadIdx.x >> 5;
    const int n0 = blockIdx.x * 64;
    const int m0 = wave * 16;
    const int lm = lane & 15;
    const int kh = lane >> 4;

    v8f acc[4] = {};
    const half_t* arow = A + (size_t)(m0 + lm) * SS_H;

    for (int k0 = 0; k0 < SS_H; k0 += 32) {
        v16h afrag = ss_load_afrag(arow + k0 + kh * 16);
#pragma unroll
        for (int t = 0; t < 4; ++t) {
            const half_t* bp = W + (size_t)(n0 + t * 16 + lm) * SS_H + k0 + kh * 16;
            v16h bfrag = *(const v16h*)bp;
            acc[t] = __builtin_amdgcn_wmma_f32_16x16x32_f16(
                false, afrag, false, bfrag, (short)0, acc[t], false, false);
        }
    }
#pragma unroll
    for (int t = 0; t < 4; ++t) {
        int n = n0 + t * 16 + lm;
#pragma unroll
        for (int i = 0; i < 8; ++i) {
            int m = m0 + kh * 8 + i;
            C[(size_t)m * N + n] = acc[t][i];
        }
    }
}

// ---------------------------------------------------------------------------
// Helpers: conversion / gather / init
// ---------------------------------------------------------------------------
__global__ void ss_convert_rows_f16(const float* __restrict__ src,
                                    half_t* __restrict__ dst,
                                    int rows, int cols, int src_stride, int col_off)
{
    size_t n = (size_t)rows * cols;
    for (size_t i = (size_t)blockIdx.x * blockDim.x + threadIdx.x; i < n;
         i += (size_t)gridDim.x * blockDim.x) {
        int r = (int)(i / cols);
        int c = (int)(i % cols);
        dst[i] = (half_t)src[(size_t)r * src_stride + col_off + c];
    }
}

__global__ void ss_gather_emb_f16(const float* __restrict__ emb,
                                  const int* __restrict__ idx,
                                  half_t* __restrict__ out, int cols)
{
    int row = blockIdx.x;
    int tok = idx[row];
    const float* s = emb + (size_t)tok * cols;
    for (int c = threadIdx.x; c < cols; c += blockDim.x)
        out[(size_t)row * cols + c] = (half_t)s[c];
}

__global__ void ss_zero_h(float* __restrict__ h, half_t* __restrict__ hf, int n)
{
    for (int i = blockIdx.x * blockDim.x + threadIdx.x; i < n;
         i += gridDim.x * blockDim.x) { h[i] = 0.0f; hf[i] = (half_t)0.0f; }
}

__device__ __forceinline__ float ss_sigmoid(float x) {
    return 1.0f / (1.0f + expf(-x));
}

// ---------------------------------------------------------------------------
// Encoder GRU gate (pointwise): h' = (1-z)*n + z*h ; also emits enc_out, h_f16
// ---------------------------------------------------------------------------
__global__ __launch_bounds__(512) void ss_gru_gate_enc(
    const float* __restrict__ gi,    // [B*S, 3H] (bih included)
    const float* __restrict__ gh,    // [B, 3H]
    const float* __restrict__ bhh,   // [3H]
    float* __restrict__ h,           // [B, H] in/out
    half_t* __restrict__ h_f16,      // [B, H]
    float* __restrict__ enc_out,     // [B, S, H]
    int t)
{
    int b = blockIdx.x, j = threadIdx.x;
    const float* gir = gi + ((size_t)b * SS_S + t) * SS_G;
    const float* ghr = gh + (size_t)b * SS_G;
    float hr = ghr[j]            + bhh[j];
    float hz = ghr[SS_H + j]     + bhh[SS_H + j];
    float hn = ghr[2 * SS_H + j] + bhh[2 * SS_H + j];
    float r = ss_sigmoid(gir[j] + hr);
    float z = ss_sigmoid(gir[SS_H + j] + hz);
    float n = tanhf(gir[2 * SS_H + j] + r * hn);
    float ho = h[b * SS_H + j];
    float hv = (1.0f - z) * n + z * ho;
    h[b * SS_H + j] = hv;
    h_f16[b * SS_H + j] = (half_t)hv;
    enc_out[((size_t)b * SS_S + t) * SS_H + j] = hv;
}

// ---------------------------------------------------------------------------
// Decoder attention: scores -> masked softmax -> context; emits ctx (f32/f16)
// and the context half of pred. One block per batch element.
// ---------------------------------------------------------------------------
__global__ __launch_bounds__(256) void ss_dec_attn(
    const float* __restrict__ enc_out,  // [B, S, H]
    const float* __restrict__ h,        // [B, H]
    const int* __restrict__ src,        // [B, S]
    float* __restrict__ ctx,            // [B, H]
    half_t* __restrict__ ctx_f16,       // [B, H]
    half_t* __restrict__ pred,          // [B*T, 2H]
    int t)
{
    __shared__ float hs[SS_H];
    __shared__ float sattn[SS_S];
    __shared__ float red[256];
    int b = blockIdx.x, tid = threadIdx.x;
    hs[tid]       = h[b * SS_H + tid];
    hs[256 + tid] = h[b * SS_H + 256 + tid];
    __syncthreads();

    const float* eb = enc_out + (size_t)b * SS_S * SS_H;
    const float* er = eb + (size_t)tid * SS_H;
    float sc = 0.0f;
    for (int k = 0; k < SS_H; ++k) sc = fmaf(er[k], hs[k], sc);
    if (src[b * SS_S + tid] == 0) sc = -1e9f;   // PAD mask

    red[tid] = sc; __syncthreads();
    for (int off = 128; off > 0; off >>= 1) {
        if (tid < off) red[tid] = fmaxf(red[tid], red[tid + off]);
        __syncthreads();
    }
    float mx = red[0]; __syncthreads();
    float ex = expf(sc - mx);
    sattn[tid] = ex;
    red[tid] = ex; __syncthreads();
    for (int off = 128; off > 0; off >>= 1) {
        if (tid < off) red[tid] += red[tid + off];
        __syncthreads();
    }
    float inv = 1.0f / red[0];

    for (int jj = tid; jj < SS_H; jj += 256) {
        float c = 0.0f;
        for (int s = 0; s < SS_S; ++s)
            c = fmaf(sattn[s] * inv, eb[(size_t)s * SS_H + jj], c);
        ctx[b * SS_H + jj] = c;
        half_t ch = (half_t)c;
        ctx_f16[b * SS_H + jj] = ch;
        pred[((size_t)b * SS_T + t) * (2 * SS_H) + SS_H + jj] = ch;
    }
}

// ---------------------------------------------------------------------------
// Decoder GRU gate (pointwise); emits h' half of pred.
// ---------------------------------------------------------------------------
__global__ __launch_bounds__(512) void ss_gru_gate_dec(
    const float* __restrict__ gie,   // [B*T, 3H] (bih included)
    const float* __restrict__ gic,   // [B, 3H]   (context part)
    const float* __restrict__ gh,    // [B, 3H]
    const float* __restrict__ bhh,   // [3H]
    float* __restrict__ h,           // [B, H] in/out
    half_t* __restrict__ h_f16,      // [B, H]
    half_t* __restrict__ pred,       // [B*T, 2H]
    int t)
{
    int b = blockIdx.x, j = threadIdx.x;
    const float* gir = gie + ((size_t)b * SS_T + t) * SS_G;
    const float* gcr = gic + (size_t)b * SS_G;
    const float* ghr = gh  + (size_t)b * SS_G;
    float ir = gir[j]            + gcr[j];
    float iz = gir[SS_H + j]     + gcr[SS_H + j];
    float in = gir[2 * SS_H + j] + gcr[2 * SS_H + j];
    float hr = ghr[j]            + bhh[j];
    float hz = ghr[SS_H + j]     + bhh[SS_H + j];
    float hn = ghr[2 * SS_H + j] + bhh[2 * SS_H + j];
    float r = ss_sigmoid(ir + hr);
    float z = ss_sigmoid(iz + hz);
    float n = tanhf(in + r * hn);
    float ho = h[b * SS_H + j];
    float hv = (1.0f - z) * n + z * ho;
    h[b * SS_H + j] = hv;
    h_f16[b * SS_H + j] = (half_t)hv;
    pred[((size_t)b * SS_T + t) * (2 * SS_H) + j] = (half_t)hv;
}

// ---------------------------------------------------------------------------
// Host-side orchestration
// ---------------------------------------------------------------------------
static inline char* ss_alloc(char*& p, size_t bytes) {
    char* r = p;
    p += (bytes + 255) & ~(size_t)255;
    return r;
}

extern "C" void kernel_launch(void* const* d_in, const int* in_sizes, int n_in,
                              void* d_out, int out_size, void* d_ws, size_t ws_size,
                              hipStream_t stream) {
    (void)in_sizes; (void)n_in; (void)out_size; (void)ws_size;
    const int*   src      = (const int*)  d_in[0];
    // d_in[1] = src_lens: unused (reference masks via src != PAD)
    const int*   tgt_in   = (const int*)  d_in[2];
    const float* enc_emb  = (const float*)d_in[3];
    const float* enc_Wih  = (const float*)d_in[4];
    const float* enc_Whh  = (const float*)d_in[5];
    const float* enc_bih  = (const float*)d_in[6];
    const float* enc_bhh  = (const float*)d_in[7];
    const float* dec_emb  = (const float*)d_in[8];
    const float* dec_Wih  = (const float*)d_in[9];   // [3H, E+H]
    const float* dec_Whh  = (const float*)d_in[10];
    const float* dec_bih  = (const float*)d_in[11];
    const float* dec_bhh  = (const float*)d_in[12];
    const float* fc_W     = (const float*)d_in[13];  // [V, 2H]
    const float* fc_b     = (const float*)d_in[14];
    float*       out      = (float*)d_out;           // [B, T, V]

    char* p = (char*)d_ws;
    half_t* emb_f16     = (half_t*)ss_alloc(p, (size_t)SS_B*SS_S*SS_E*2);      // 4 MB
    half_t* encWih_f16  = (half_t*)ss_alloc(p, (size_t)SS_G*SS_E*2);
    half_t* encWhh_f16  = (half_t*)ss_alloc(p, (size_t)SS_G*SS_H*2);
    float*  enc_gi      = (float*) ss_alloc(p, (size_t)SS_B*SS_S*SS_G*4);      // 50 MB
    float*  enc_out     = (float*) ss_alloc(p, (size_t)SS_B*SS_S*SS_H*4);      // 17 MB
    float*  h_st        = (float*) ss_alloc(p, (size_t)SS_B*SS_H*4);
    half_t* h_f16       = (half_t*)ss_alloc(p, (size_t)SS_B*SS_H*2);
    float*  gh_buf      = (float*) ss_alloc(p, (size_t)SS_B*SS_G*4);
    float*  gic_buf     = (float*) ss_alloc(p, (size_t)SS_B*SS_G*4);
    half_t* demb_f16    = (half_t*)ss_alloc(p, (size_t)SS_B*SS_T*SS_E*2);
    half_t* decWihE_f16 = (half_t*)ss_alloc(p, (size_t)SS_G*SS_E*2);
    half_t* decWihC_f16 = (half_t*)ss_alloc(p, (size_t)SS_G*SS_H*2);
    half_t* decWhh_f16  = (half_t*)ss_alloc(p, (size_t)SS_G*SS_H*2);
    float*  dec_gie     = (float*) ss_alloc(p, (size_t)SS_B*SS_T*SS_G*4);      // 13 MB
    float*  ctx_f32     = (float*) ss_alloc(p, (size_t)SS_B*SS_H*4);
    half_t* ctx_f16     = (half_t*)ss_alloc(p, (size_t)SS_B*SS_H*2);
    half_t* pred_f16    = (half_t*)ss_alloc(p, (size_t)SS_B*SS_T*2*SS_H*2);    // 4 MB
    half_t* fcW_f16     = (half_t*)ss_alloc(p, (size_t)SS_V*2*SS_H*2);         // 66 MB

    // --- weight conversions to f16 (K-contiguous row-major) ---
    auto cgrid = [](size_t n) { return dim3((unsigned)((n + 255) / 256)); };
    ss_convert_rows_f16<<<cgrid((size_t)SS_G*SS_E), 256, 0, stream>>>(enc_Wih, encWih_f16, SS_G, SS_E, SS_E, 0);
    ss_convert_rows_f16<<<cgrid((size_t)SS_G*SS_H), 256, 0, stream>>>(enc_Whh, encWhh_f16, SS_G, SS_H, SS_H, 0);
    ss_convert_rows_f16<<<cgrid((size_t)SS_G*SS_E), 256, 0, stream>>>(dec_Wih, decWihE_f16, SS_G, SS_E, SS_E + SS_H, 0);
    ss_convert_rows_f16<<<cgrid((size_t)SS_G*SS_H), 256, 0, stream>>>(dec_Wih, decWihC_f16, SS_G, SS_H, SS_E + SS_H, SS_E);
    ss_convert_rows_f16<<<cgrid((size_t)SS_G*SS_H), 256, 0, stream>>>(dec_Whh, decWhh_f16, SS_G, SS_H, SS_H, 0);
    ss_convert_rows_f16<<<cgrid((size_t)SS_V*2*SS_H), 256, 0, stream>>>(fc_W, fcW_f16, SS_V, 2*SS_H, 2*SS_H, 0);

    // --- embedding gathers (f16) ---
    ss_gather_emb_f16<<<SS_B*SS_S, 256, 0, stream>>>(enc_emb, src, emb_f16, SS_E);
    ss_gather_emb_f16<<<SS_B*SS_T, 256, 0, stream>>>(dec_emb, tgt_in, demb_f16, SS_E);

    // --- hoisted batched GEMMs (WMMA) ---
    ss_gemm_wmma<<<dim3(SS_G/64, (SS_B*SS_S)/128), 128, 0, stream>>>(
        emb_f16, encWih_f16, enc_bih, enc_gi, SS_B*SS_S, SS_G, SS_E);
    ss_gemm_wmma<<<dim3(SS_G/64, (SS_B*SS_T)/128), 128, 0, stream>>>(
        demb_f16, decWihE_f16, dec_bih, dec_gie, SS_B*SS_T, SS_G, SS_E);

    // --- init hidden state ---
    ss_zero_h<<<64, 256, 0, stream>>>(h_st, h_f16, SS_B*SS_H);

    // --- encoder scan: gh = h @ Whh^T (WMMA), then pointwise gates ---
    for (int t = 0; t < SS_S; ++t) {
        ss_gemm32_wmma<<<dim3(SS_G/64), 64, 0, stream>>>(h_f16, encWhh_f16, gh_buf, SS_G);
        ss_gru_gate_enc<<<SS_B, SS_H, 0, stream>>>(enc_gi, gh_buf, enc_bhh,
                                                   h_st, h_f16, enc_out, t);
    }

    // --- decoder scan: attention -> two small WMMA GEMMs -> gates ---
    for (int t = 0; t < SS_T; ++t) {
        ss_dec_attn<<<SS_B, 256, 0, stream>>>(enc_out, h_st, src,
                                              ctx_f32, ctx_f16, pred_f16, t);
        ss_gemm32_wmma<<<dim3(SS_G/64), 64, 0, stream>>>(h_f16,   decWhh_f16,  gh_buf,  SS_G);
        ss_gemm32_wmma<<<dim3(SS_G/64), 64, 0, stream>>>(ctx_f16, decWihC_f16, gic_buf, SS_G);
        ss_gru_gate_dec<<<SS_B, SS_H, 0, stream>>>(dec_gie, gic_buf, gh_buf, dec_bhh,
                                                   h_st, h_f16, pred_f16, t);
    }

    // --- dominant GEMM: logits = pred @ fc_W^T + fc_b  [2048 x 32000] ---
    ss_gemm_wmma<<<dim3(SS_V/64, (SS_B*SS_T)/128), 128, 0, stream>>>(
        pred_f16, fcW_f16, fc_b, out, SS_B*SS_T, SS_V, 2*SS_H);
}
